// VisualEncoder_89799176225230
// MI455X (gfx1250) — matrix operand order repaired and checked by reading, not compiled, not run on hardware
//
#include <hip/hip_runtime.h>
#include <hip/hip_bf16.h>
#include <math.h>

// ---------------------------------------------------------------------------
// CDNA5 / gfx1250 wave32 WMMA types
// ---------------------------------------------------------------------------
typedef __attribute__((ext_vector_type(16))) __bf16 v16bf;
typedef __attribute__((ext_vector_type(8)))  float  v8f;

#define WMMA_BF16(a, b, c) \
  __builtin_amdgcn_wmma_f32_16x16x32_bf16(false, (a), false, (b), (short)0, (c), false, false)

__device__ __forceinline__ float gelu_exact(float x) {
  return 0.5f * x * (1.0f + erff(x * 0.70710678118654752f));
}

// ---------------------------------------------------------------------------
// Software-pipelined, LDS double-buffered GEMM:
//   C[M,Nc] = A[M,K] * W[Nc,K]^T (+bias)(gelu)(+R)
// Block tile 128x64 (8 waves, each wave 16 rows x 64 cols), BK=32.
// Pipeline per k-step: (1) issue next tile's global_load_b128s (no wait),
// (2) ds_load all fragments + 4 back-to-back WMMAs (hides global latency),
// (3) wait loads, cvt f32->bf16 once per block, ds_store next buffer.
// LDS rows padded to 40 bf16 (80B) -> 16 fragment rows hit distinct banks.
// flags: bit0 = gelu, bit1 = residual add (R).
// ---------------------------------------------------------------------------
#define BM 128
#define BN 64
#define LDA 40   // padded row stride (bf16 elems)

__global__ __launch_bounds__(256)
void gemm_bf16_wmma(const float* __restrict__ A, const float* __restrict__ W,
                    const float* __restrict__ bias, const float* __restrict__ R,
                    float* __restrict__ C, int M, int Nc, int K, int flags)
{
  __shared__ __bf16 As[2][BM * LDA];
  __shared__ __bf16 Ws[2][BN * LDA];

  const int tid  = threadIdx.x;
  const int lane = tid & 31;
  const int wv   = tid >> 5;            // wave 0..7 -> row strip
  const int l16  = lane & 15;
  const int half = lane >> 4;

  const int mb  = (M + BM - 1) / BM;
  const int bm0 = (blockIdx.x % mb) * BM;
  const int bn0 = (blockIdx.x / mb) * BN;

  // staging roles: A: (row = tid>>1, 16-wide seg = tid&1); W: (row = tid&63, 8-wide seg = tid>>6)
  const int  arow = tid >> 1, aseg = tid & 1;
  const bool avld = (bm0 + arow) < M;
  const float* aptr = A + (size_t)(avld ? (bm0 + arow) : (M - 1)) * K + aseg * 16;
  const int  wrow = tid & 63, wseg = tid >> 6;
  const float* wptr = W + (size_t)(bn0 + wrow) * K + wseg * 8;

  float4 ra[4], rw[2];                  // in-flight staging registers

  auto load_tiles = [&](int kk) {       // issue global loads, do NOT consume
    const float4* sa = (const float4*)(aptr + kk);
#pragma unroll
    for (int i = 0; i < 4; ++i) ra[i] = sa[i];
    const float4* sw = (const float4*)(wptr + kk);
    rw[0] = sw[0]; rw[1] = sw[1];
    if (kk + 64 < K) __builtin_prefetch(wptr + kk + 64, 0, 0);
  };

  auto store_tiles = [&](int buf) {     // consume (waits), cvt, ds_store
    __bf16* da = &As[buf][arow * LDA + aseg * 16];
    const float* fa = (const float*)ra;
#pragma unroll
    for (int i = 0; i < 16; ++i) da[i] = (__bf16)(avld ? fa[i] : 0.0f);
    __bf16* dw = &Ws[buf][wrow * LDA + wseg * 8];
    const float* fw = (const float*)rw;
#pragma unroll
    for (int i = 0; i < 8; ++i) dw[i] = (__bf16)fw[i];
  };

  v8f acc0 = {}, acc1 = {}, acc2 = {}, acc3 = {};

  const int ksteps = K >> 5;
  load_tiles(0);
  store_tiles(0);
  __syncthreads();

  for (int kt = 0; kt < ksteps; ++kt) {
    const int  cbuf = kt & 1;
    const bool more = (kt + 1) < ksteps;
    if (more) load_tiles((kt + 1) << 5);        // overlap next global loads

    // all fragment ds_loads issued before the WMMA chain (DS is in-order)
    const __bf16* pa = &As[cbuf][(wv * 16 + l16) * LDA];
    const __bf16* pw = &Ws[cbuf][l16 * LDA + half * 16];
    v16bf af, bf0, bf1, bf2, bf3;
#pragma unroll
    for (int e = 0; e < 8; ++e) { af[e] = pa[half * 16 + e]; af[8 + e] = pa[half * 16 + 8 + e]; }
#pragma unroll
    for (int e = 0; e < 16; ++e) bf0[e] = pw[e];
#pragma unroll
    for (int e = 0; e < 16; ++e) bf1[e] = pw[16 * LDA + e];
#pragma unroll
    for (int e = 0; e < 16; ++e) bf2[e] = pw[32 * LDA + e];
#pragma unroll
    for (int e = 0; e < 16; ++e) bf3[e] = pw[48 * LDA + e];

    acc0 = WMMA_BF16(af, bf0, acc0);            // no D->A/B hazards: back-to-back
    acc1 = WMMA_BF16(af, bf1, acc1);
    acc2 = WMMA_BF16(af, bf2, acc2);
    acc3 = WMMA_BF16(af, bf3, acc3);

    if (more) store_tiles(cbuf ^ 1);            // loadcnt wait lands AFTER compute
    __syncthreads();
  }

  // epilogue: lane holds rows r + 8*half of its wave strip
  v8f accv[4] = {acc0, acc1, acc2, acc3};
#pragma unroll
  for (int j = 0; j < 4; ++j) {
#pragma unroll
    for (int r = 0; r < 8; ++r) {
      int m = bm0 + wv * 16 + r + 8 * half;
      int n = bn0 + j * 16 + l16;
      if (m < M) {
        float v = accv[j][r];
        if (bias) v += bias[n];
        if (flags & 1) v = gelu_exact(v);
        if (flags & 2) v += R[(size_t)m * Nc + n];
        C[(size_t)m * Nc + n] = v;
      }
    }
  }
}

// ---------------------------------------------------------------------------
// Attention for one (image, head, 16-row tile). SEQ=169 -> 11 key tiles,
// padded to 192. Block-diagonal mask handled structurally (per-image tiles).
// One wave per workgroup. QK^T and P*V via WMMA; softmax via __shfl_xor
// within 16-lane half-groups; P staged bf16 in LDS (C-layout -> A-fragment).
// All global loads use clamped pointers + lane selects (no EXEC divergence).
// ---------------------------------------------------------------------------
__global__ __launch_bounds__(32)
void attn_wmma(const float* __restrict__ qkv, float* __restrict__ out)
{
  __shared__ __bf16 P[16 * 192];
  const int lane = threadIdx.x;
  const int l16  = lane & 15;
  const int half = lane >> 4;
  const int rt  = blockIdx.x;   // row tile 0..10
  const int hd  = blockIdx.y;   // head 0..11
  const int blk = blockIdx.z;   // image 0..15
  const int base = blk * 169;

  // ---- Q fragments (A layout), rows rt*16 + l16, d = 0..63 in two k-steps
  const int  qr   = rt * 16 + l16;
  const bool qvld = qr < 169;
  const float* qp = qkv + (size_t)(base + (qvld ? qr : 168)) * 2304 + hd * 64;
  float q0[16], q1[16];
#pragma unroll
  for (int e = 0; e < 16; ++e) {
    int d = half * 16 + e;
    q0[e] = qp[d];
    q1[e] = qp[32 + d];
  }
  v16bf aq0, aq1;
#pragma unroll
  for (int e = 0; e < 16; ++e) {
    aq0[e] = (__bf16)(qvld ? q0[e] : 0.0f);
    aq1[e] = (__bf16)(qvld ? q1[e] : 0.0f);
  }

  // ---- scores S = (Q K^T) * 0.125, column-masked at 169
  float sc[11][8];
#pragma unroll
  for (int ct = 0; ct < 11; ++ct) {
    const int  kc   = ct * 16 + l16;
    const bool kvld = kc < 169;
    const float* kp = qkv + (size_t)(base + (kvld ? kc : 168)) * 2304 + 768 + hd * 64;
    float k0[16], k1[16];
#pragma unroll
    for (int e = 0; e < 16; ++e) { int d = half * 16 + e; k0[e] = kp[d]; k1[e] = kp[32 + d]; }
    v16bf bk0, bk1;
#pragma unroll
    for (int e = 0; e < 16; ++e) {
      bk0[e] = (__bf16)(kvld ? k0[e] : 0.0f);
      bk1[e] = (__bf16)(kvld ? k1[e] : 0.0f);
    }
    v8f s = {};
    s = WMMA_BF16(aq0, bk0, s);
    s = WMMA_BF16(aq1, bk1, s);
#pragma unroll
    for (int r = 0; r < 8; ++r)
      sc[ct][r] = kvld ? s[r] * 0.125f : -__builtin_inff();
  }

  // ---- softmax per row (row = r + 8*half; 16 lanes of same half share a row)
#pragma unroll
  for (int r = 0; r < 8; ++r) {
    float m = sc[0][r];
#pragma unroll
    for (int ct = 1; ct < 11; ++ct) m = fmaxf(m, sc[ct][r]);
#pragma unroll
    for (int off = 8; off >= 1; off >>= 1) m = fmaxf(m, __shfl_xor(m, off, 32));
    float su = 0.0f;
#pragma unroll
    for (int ct = 0; ct < 11; ++ct) { sc[ct][r] = __expf(sc[ct][r] - m); su += sc[ct][r]; }
#pragma unroll
    for (int off = 8; off >= 1; off >>= 1) su += __shfl_xor(su, off, 32);
    float rinv = 1.0f / su;
#pragma unroll
    for (int ct = 0; ct < 11; ++ct) sc[ct][r] *= rinv;
  }

  // ---- stage P (bf16) into LDS, padded to 192 cols with zeros
#pragma unroll
  for (int r = 0; r < 8; ++r) {
    int row = r + 8 * half;
#pragma unroll
    for (int ct = 0; ct < 11; ++ct)
      P[row * 192 + ct * 16 + l16] = (__bf16)sc[ct][r];
    P[row * 192 + 176 + l16] = (__bf16)0.0f;
  }
  __syncthreads();

  // ---- O = P * V  (K over 192 in 6 steps of 32; keys >=169 contribute P=0)
  v8f oacc0 = {}, oacc1 = {}, oacc2 = {}, oacc3 = {};
  v8f* oaccs[4] = {&oacc0, &oacc1, &oacc2, &oacc3};
#pragma unroll
  for (int kt = 0; kt < 6; ++kt) {
    v16bf ap;
#pragma unroll
    for (int e = 0; e < 16; ++e) {
      int kk = kt * 32 + half * 16 + e;
      ap[e] = P[l16 * 192 + kk];
    }
#pragma unroll
    for (int j = 0; j < 4; ++j) {
      float fv[16];
#pragma unroll
      for (int e = 0; e < 16; ++e) {
        int key = kt * 32 + half * 16 + e;
        int kcl = key < 169 ? key : 168;
        fv[e] = qkv[(size_t)(base + kcl) * 2304 + 1536 + hd * 64 + j * 16 + l16];
      }
      v16bf bv;
#pragma unroll
      for (int e = 0; e < 16; ++e) {
        int key = kt * 32 + half * 16 + e;
        bv[e] = (__bf16)(key < 169 ? fv[e] : 0.0f);
      }
      *oaccs[j] = WMMA_BF16(ap, bv, *oaccs[j]);
    }
  }
  v8f oaccv[4] = {oacc0, oacc1, oacc2, oacc3};
#pragma unroll
  for (int j = 0; j < 4; ++j)
#pragma unroll
    for (int r = 0; r < 8; ++r) {
      int m = rt * 16 + r + 8 * half;
      if (m < 169)
        out[(size_t)(base + m) * 768 + hd * 64 + j * 16 + l16] = oaccv[j][r];
    }
}

// ---------------------------------------------------------------------------
// Row LayerNorm: one block per row, E up to 2048
// ---------------------------------------------------------------------------
__global__ __launch_bounds__(256)
void layernorm(const float* __restrict__ X, const float* __restrict__ g,
               const float* __restrict__ b, float* __restrict__ Y, int E)
{
  __shared__ float rs[256], rs2[256];
  const int row = blockIdx.x, tid = threadIdx.x;
  const float* x = X + (size_t)row * E;
  float s = 0.f, s2 = 0.f;
  for (int i = tid; i < E; i += 256) { float v = x[i]; s += v; s2 += v * v; }
  rs[tid] = s; rs2[tid] = s2; __syncthreads();
  for (int off = 128; off > 0; off >>= 1) {
    if (tid < off) { rs[tid] += rs[tid + off]; rs2[tid] += rs2[tid + off]; }
    __syncthreads();
  }
  float mean = rs[0] / E;
  float var  = rs2[0] / E - mean * mean;
  float rstd = rsqrtf(var + 1e-5f);
  float* y = Y + (size_t)row * E;
  for (int i = tid; i < E; i += 256) y[i] = (x[i] - mean) * rstd * g[i] + b[i];
}

// ---------------------------------------------------------------------------
// Patchify: [16,3,64,672] -> [2688, 768] rows of (c,py,px)
// ---------------------------------------------------------------------------
__global__ void patchify(const float* __restrict__ px, float* __restrict__ xp)
{
  int idx = blockIdx.x * 256 + threadIdx.x;
  if (idx >= 2688 * 768) return;
  int e = idx % 768, p = idx / 768;
  int b = p / 168, pp = p % 168;
  int ph = pp / 42, pw = pp % 42;
  int c = e / 256, rem = e % 256, py = rem / 16, pxx = rem % 16;
  xp[idx] = px[(((size_t)b * 3 + c) * 64 + ph * 16 + py) * 672 + pw * 16 + pxx];
}

// ---------------------------------------------------------------------------
// Assemble sequence: CLS + (patch_embed + 2D pos) -> x0 [2704, 768]
// ---------------------------------------------------------------------------
__global__ void assemble_seq(const float* __restrict__ pe, const float* __restrict__ cls,
                             const float* __restrict__ row_emb, const float* __restrict__ col_emb,
                             float* __restrict__ x0)
{
  int idx = blockIdx.x * 256 + threadIdx.x;
  if (idx >= 2704 * 768) return;
  int e = idx % 768, n = idx / 768;
  int b = n / 169, s = n % 169;
  float v;
  if (s == 0) v = cls[e];
  else {
    int p = s - 1;
    float pos = (e < 384) ? row_emb[(p / 42) * 384 + e]
                          : col_emb[(p % 42) * 384 + (e - 384)];
    v = pe[((size_t)b * 168 + p) * 768 + e] + pos;
  }
  x0[idx] = v;
}

__global__ void set_mask(int* __restrict__ m)
{
  int i = blockIdx.x * 256 + threadIdx.x;
  if (i < 2704) m[i] = 1;
}

// ---------------------------------------------------------------------------
// Host orchestration
// ---------------------------------------------------------------------------
extern "C" void kernel_launch(void* const* d_in, const int* in_sizes, int n_in,
                              void* d_out, int out_size, void* d_ws, size_t ws_size,
                              hipStream_t stream)
{
  const float* pixel   = (const float*)d_in[0];
  const float* patch_W = (const float*)d_in[1];
  const float* patch_b = (const float*)d_in[2];
  const float* row_emb = (const float*)d_in[3];
  const float* col_emb = (const float*)d_in[4];
  const float* cls     = (const float*)d_in[5];
  const float* ln1_g   = (const float*)d_in[6];
  const float* ln1_b   = (const float*)d_in[7];
  const float* in_w    = (const float*)d_in[8];
  const float* in_b    = (const float*)d_in[9];
  const float* out_w   = (const float*)d_in[10];
  const float* out_b   = (const float*)d_in[11];
  const float* ln2_g   = (const float*)d_in[12];
  const float* ln2_b   = (const float*)d_in[13];
  const float* mlp_w1  = (const float*)d_in[14];
  const float* mlp_b1  = (const float*)d_in[15];
  const float* mlp_w2  = (const float*)d_in[16];
  const float* mlp_b2  = (const float*)d_in[17];
  const float* fin_g   = (const float*)d_in[18];
  const float* fin_b   = (const float*)d_in[19];
  const float* br_w1   = (const float*)d_in[20];
  const float* br_b1   = (const float*)d_in[21];
  const float* br_w2   = (const float*)d_in[22];
  const float* br_b2   = (const float*)d_in[23];
  const float* br_g    = (const float*)d_in[24];
  const float* br_b    = (const float*)d_in[25];

  // workspace layout (floats)
  float* ws   = (float*)d_ws;
  float* x    = ws;                      // 2704*768
  float* xn   = x    + 2076672;          // 2704*768
  float* qkv  = xn   + 2076672;          // 2704*2304
  float* attn = qkv  + 6230016;          // 2704*768
  float* h1   = attn + 2076672;          // 2704*3072
  float* xpatch = attn;                  // reuse: 2688*768 (pre-layer only)
  float* pe     = h1;                    // reuse: 2688*768 (pre-layer only)
  float* b1     = h1;                    // reuse: 2704*1536 (post-layer)
  float* b2     = qkv;                   // reuse: 2704*2048 (post-layer)

  auto gemm = [&](const float* A, const float* W, const float* bias, const float* R,
                  float* C, int M, int Nc, int K, int flags) {
    int mb = (M + BM - 1) / BM;
    int nb = Nc / BN;
    gemm_bf16_wmma<<<mb * nb, 256, 0, stream>>>(A, W, bias, R, C, M, Nc, K, flags);
  };

  // patch embedding
  patchify<<<(2688 * 768 + 255) / 256, 256, 0, stream>>>(pixel, xpatch);
  gemm(xpatch, patch_W, patch_b, nullptr, pe, 2688, 768, 768, 0);
  assemble_seq<<<(2704 * 768 + 255) / 256, 256, 0, stream>>>(pe, cls, row_emb, col_emb, x);

  // 12 transformer layers
  for (int l = 0; l < 12; ++l) {
    layernorm<<<2704, 256, 0, stream>>>(x, ln1_g + l * 768, ln1_b + l * 768, xn, 768);
    gemm(xn, in_w + (size_t)l * 2304 * 768, in_b + l * 2304, nullptr, qkv, 2704, 2304, 768, 0);
    attn_wmma<<<dim3(11, 12, 16), 32, 0, stream>>>(qkv, attn);
    gemm(attn, out_w + (size_t)l * 768 * 768, out_b + l * 768, x, x, 2704, 768, 768, 2);
    layernorm<<<2704, 256, 0, stream>>>(x, ln2_g + l * 768, ln2_b + l * 768, xn, 768);
    gemm(xn, mlp_w1 + (size_t)l * 3072 * 768, mlp_b1 + l * 3072, nullptr, h1, 2704, 3072, 768, 1);
    gemm(h1, mlp_w2 + (size_t)l * 768 * 3072, mlp_b2 + l * 768, x, x, 2704, 768, 3072, 2);
  }

  // final LN + branch head
  layernorm<<<2704, 256, 0, stream>>>(x, fin_g, fin_b, xn, 768);
  gemm(xn, br_w1, br_b1, nullptr, b1, 2704, 1536, 768, 1);
  gemm(b1, br_w2, br_b2, nullptr, b2, 2704, 2048, 1536, 0);
  layernorm<<<2704, 256, 0, stream>>>(b2, br_g, br_b, (float*)d_out, 2048);

  // vis_mask (int32 ones) appended after the [2704,2048] f32 output
  set_mask<<<(2704 + 255) / 256, 256, 0, stream>>>(
      (int*)((float*)d_out + (size_t)2704 * 2048));
}
